// MultiHeadAttentionLayer_65575560675527
// MI455X (gfx1250) — matrix operand order repaired and checked by reading, not compiled, not run on hardware
//
#include <hip/hip_runtime.h>
#include <math.h>

#define BATCH 4
#define SEQ   1024
#define HIDN  1024
#define NH    16
#define HD    64

#define BM 128
#define BN 64
#define BK 32

typedef _Float16 v16h __attribute__((ext_vector_type(16)));
typedef _Float16 h8   __attribute__((ext_vector_type(8)));
typedef _Float16 h4   __attribute__((ext_vector_type(4)));
typedef float    v8f  __attribute__((ext_vector_type(8)));

// ---------------------------------------------------------------------------
// Core WMMA GEMM tile: C[rowBase:+128, colBase:+64] = scale*(A@B(^T)) (+bias)
// A: MxK row-major f32 (lda), B: KxN row-major (NN) or NxK row-major (NT).
// f32 -> f16 conversion while staging to LDS (packed ds_store_b64);
// accumulate in f32 with v_wmma_f32_16x16x32_f16.
// 256 threads = 8 waves; each wave owns a 32x32 sub-tile (2x2 WMMA frags).
// ---------------------------------------------------------------------------
template <bool TRANS_B, bool HAS_BIAS>
__device__ __forceinline__ void gemm_tile(const float* __restrict__ A, int lda,
                                          const float* __restrict__ Bm, int ldb,
                                          const float* __restrict__ bias, float scale,
                                          float* __restrict__ C, int ldc,
                                          int K, int rowBase, int colBase)
{
    // +8 half padding -> row stride 40 halfs = 80 B (16B multiple: aligned b128
    // fragment loads, staggered banks)
    __shared__ _Float16 As[BM][BK + 8];
    __shared__ _Float16 Bs[BN][BK + 8];   // stored N-major: Bs[n][k]

    const int tid  = threadIdx.x;
    const int lane = tid & 31;
    const int wid  = tid >> 5;
    const int wm   = wid & 3;   // 4 waves along M (32 rows each)
    const int wn   = wid >> 2;  // 2 waves along N (32 cols each)

    v8f acc[2][2] = {};

    for (int kb = 0; kb < K; kb += BK) {
        __syncthreads();
        // ---- stage A tile: 128x32 f32 -> f16, 4x (float4 load + b64 LDS store) ----
#pragma unroll
        for (int i = 0; i < 4; ++i) {
            const int idx = tid * 4 + i;        // 0..1023 float4s
            const int r   = idx >> 3;           // 8 float4 per 32-float row
            const int c   = (idx & 7) * 4;
            const float4 f = *(const float4*)(A + (size_t)(rowBase + r) * lda + kb + c);
            const h4 hv = { (_Float16)f.x, (_Float16)f.y, (_Float16)f.z, (_Float16)f.w };
            *(h4*)&As[r][c] = hv;               // 8B aligned (80B row stride)
        }
        // ---- stage B tile into N-major LDS ----
        if (TRANS_B) {
            // B is NxK row-major: straight row copy along K
#pragma unroll
            for (int i = 0; i < 2; ++i) {
                const int idx = tid * 2 + i;    // 0..511 float4s
                const int n   = idx >> 3;
                const int c   = (idx & 7) * 4;
                const float4 f = *(const float4*)(Bm + (size_t)(colBase + n) * ldb + kb + c);
                const h4 hv = { (_Float16)f.x, (_Float16)f.y, (_Float16)f.z, (_Float16)f.w };
                *(h4*)&Bs[n][c] = hv;
            }
        } else {
            // B is KxN row-major: each thread gathers 4 consecutive K for one
            // column (adjacent lanes -> adjacent columns: coalesced), then one
            // packed b64 LDS store into the N-major tile.
            const int n = tid & 63;
#pragma unroll
            for (int i = 0; i < 2; ++i) {
                const int kq = (tid >> 6) + i * 4;   // 0..7 (quads of K)
                const size_t gb = (size_t)(kb + kq * 4) * ldb + colBase + n;
                const float f0 = Bm[gb];
                const float f1 = Bm[gb + (size_t)ldb];
                const float f2 = Bm[gb + 2 * (size_t)ldb];
                const float f3 = Bm[gb + 3 * (size_t)ldb];
                const h4 hv = { (_Float16)f0, (_Float16)f1, (_Float16)f2, (_Float16)f3 };
                *(h4*)&Bs[n][kq * 4] = hv;
            }
        }
        __syncthreads();

        // ---- build fragments (ISA 16-bit A 16x32 / B 32x16 layouts) ----
        const int mn  = lane & 15;
        const int akl = (lane < 16) ? 0 : 8;    // A: lanes<16 -> K 0..7,16..23
        const int bkl = (lane < 16) ? 0 : 8;    // B: lanes<16 -> K 0..7,16..23

        v16h af[2], bf[2];
#pragma unroll
        for (int sm = 0; sm < 2; ++sm) {
            const int m = wm * 32 + sm * 16 + mn;
            const h8 lo = *(const h8*)&As[m][akl];
            const h8 hi = *(const h8*)&As[m][akl + 16];
#pragma unroll
            for (int j = 0; j < 8; ++j) { af[sm][j] = lo[j]; af[sm][8 + j] = hi[j]; }
        }
#pragma unroll
        for (int sn = 0; sn < 2; ++sn) {
            const int n = wn * 32 + sn * 16 + mn;
            const h8 lo = *(const h8*)&Bs[n][bkl];
            const h8 hi = *(const h8*)&Bs[n][bkl + 16];
#pragma unroll
            for (int j = 0; j < 8; ++j) { bf[sn][j] = lo[j]; bf[sn][8 + j] = hi[j]; }
        }
#pragma unroll
        for (int sm = 0; sm < 2; ++sm)
#pragma unroll
            for (int sn = 0; sn < 2; ++sn)
                acc[sm][sn] = __builtin_amdgcn_wmma_f32_16x16x32_f16(
                    false, af[sm], false, bf[sn], (short)0, acc[sm][sn], false, false);
    }

    // ---- branch-free epilogue (C layout: VGPR r -> M=r / M=8+r upper lanes) ----
    float bcol[2];
#pragma unroll
    for (int sn = 0; sn < 2; ++sn)
        bcol[sn] = HAS_BIAS ? bias[colBase + wn * 32 + sn * 16 + (lane & 15)] : 0.0f;

    const int radd = (lane >> 4) << 3;
#pragma unroll
    for (int sm = 0; sm < 2; ++sm) {
        const int row0 = rowBase + wm * 32 + sm * 16 + radd;
#pragma unroll
        for (int sn = 0; sn < 2; ++sn) {
            const int col = colBase + wn * 32 + sn * 16 + (lane & 15);
            float* cp = C + (size_t)row0 * ldc + col;
#pragma unroll
            for (int r = 0; r < 8; ++r)
                cp[(size_t)r * ldc] = acc[sm][sn][r] * scale + bcol[sn];
        }
    }
}

// ---------------------------------------------------------------------------
__global__ __launch_bounds__(256)
void proj_gemm(const float* __restrict__ A, const float* __restrict__ Bm,
               const float* __restrict__ bias, float* __restrict__ C,
               int K, int lda, int ldb, int ldc)
{
    gemm_tile<false, true>(A, lda, Bm, ldb, bias, 1.0f, C, ldc, K,
                           blockIdx.y * BM, blockIdx.x * BN);
}

__global__ __launch_bounds__(256)
void scores_gemm(const float* __restrict__ q, const float* __restrict__ k,
                 float* __restrict__ attn)
{
    const int z = blockIdx.z, h = z & (NH - 1), b = z >> 4;
    const float* qp = q + (size_t)b * SEQ * HIDN + h * HD;
    const float* kp = k + (size_t)b * SEQ * HIDN + h * HD;
    float* cp = attn + (size_t)z * SEQ * SEQ;
    gemm_tile<true, false>(qp, HIDN, kp, HIDN, nullptr, 0.125f /*1/sqrt(64)*/,
                           cp, SEQ, HD, blockIdx.y * BM, blockIdx.x * BN);
}

__global__ __launch_bounds__(256)
void attnv_gemm(const float* __restrict__ attn, const float* __restrict__ v,
                float* __restrict__ outm)
{
    const int z = blockIdx.z, h = z & (NH - 1), b = z >> 4;
    const float* ap = attn + (size_t)z * SEQ * SEQ;
    const float* vp = v + (size_t)b * SEQ * HIDN + h * HD;
    float* cp = outm + (size_t)b * SEQ * HIDN + h * HD;
    gemm_tile<false, false>(ap, SEQ, vp, HIDN, nullptr, 1.0f, cp, HIDN,
                            SEQ, blockIdx.y * BM, blockIdx.x * BN);
}

// ---------------------------------------------------------------------------
// Gaussian-window params: per (b,h,s) wave: tp = tanh(p[.,h*64:+64]),
// cen = S*sigmoid(tp.Wup + bup), win = S*sigmoid(tp.Wud + bud)
// ---------------------------------------------------------------------------
__global__ __launch_bounds__(256)
void gauss_window_params(const float* __restrict__ p,
                         const float* __restrict__ Wup, const float* __restrict__ bup,
                         const float* __restrict__ Wud, const float* __restrict__ bud,
                         float* __restrict__ cen, float* __restrict__ win)
{
    const int lane = threadIdx.x & 31;
    const int gw   = blockIdx.x * 8 + (threadIdx.x >> 5);  // (b*NH+h)*SEQ + s
    const int s  = gw & (SEQ - 1);
    const int bh = gw >> 10;
    const int h  = bh & (NH - 1);
    const int b  = bh >> 4;
    const float* prow = p + ((size_t)(b * SEQ + s)) * HIDN + h * HD;
    const float t0 = tanhf(prow[lane]);
    const float t1 = tanhf(prow[lane + 32]);
    float pi = t0 * Wup[lane] + t1 * Wup[lane + 32];
    float zi = t0 * Wud[lane] + t1 * Wud[lane + 32];
#pragma unroll
    for (int off = 16; off > 0; off >>= 1) {
        pi += __shfl_xor(pi, off, 32);
        zi += __shfl_xor(zi, off, 32);
    }
    if (lane == 0) {
        pi += bup[0];
        zi += bud[0];
        cen[gw] = (float)SEQ / (1.0f + __expf(-pi));
        win[gw] = (float)SEQ / (1.0f + __expf(-zi));
    }
}

// alpha[b,h,s] = sigmoid(query[b,s,:] . Wus[:,h] + bus[h])  (stored [B,NH,S])
__global__ __launch_bounds__(256)
void alpha_gate(const float* __restrict__ query,
                const float* __restrict__ Wus, const float* __restrict__ bus,
                float* __restrict__ alT)
{
    const int lane = threadIdx.x & 31;
    const int gw   = blockIdx.x * 8 + (threadIdx.x >> 5);  // (b*SEQ+s)*NH + h
    const int h  = gw & (NH - 1);
    const int bs = gw >> 4;
    const float* qrow = query + (size_t)bs * HIDN;
    float acc = 0.f;
    for (int i = lane; i < HIDN; i += 32) acc += qrow[i] * Wus[i * NH + h];
#pragma unroll
    for (int off = 16; off > 0; off >>= 1) acc += __shfl_xor(acc, off, 32);
    if (lane == 0) {
        const int b = bs >> 10, s2 = bs & (SEQ - 1);
        alT[((size_t)(b * NH + h)) * SEQ + s2] = 1.0f / (1.0f + __expf(-(acc + bus[h])));
    }
}

// ---------------------------------------------------------------------------
// Fused gated-Gaussian-bias + softmax, in place over one 1024-wide row.
// Folding the bias here avoids an extra full pass over the 268 MB attn tensor.
// ---------------------------------------------------------------------------
__global__ __launch_bounds__(256)
void softmax_bias_rows(float* __restrict__ attn,
                       const float* __restrict__ cen,
                       const float* __restrict__ win,
                       const float* __restrict__ alT)
{
    const int row = blockIdx.x;            // (b*NH+h)*SEQ + q
    const float c    = cen[row];
    const float w    = win[row];
    const float al   = alT[row];
    const float coef = al / (0.5f * w * w);
    float* x = attn + (size_t)row * SEQ;
    const int t = threadIdx.x;

    float v[4];
    float mx = -3.402823466e38f;
#pragma unroll
    for (int i = 0; i < 4; ++i) {
        const int col = t + i * 256;
        const float d = (float)col - c;
        v[i] = x[col] - coef * d * d;
        mx = fmaxf(mx, v[i]);
    }
    __shared__ float red[8];
#pragma unroll
    for (int off = 16; off > 0; off >>= 1) mx = fmaxf(mx, __shfl_xor(mx, off, 32));
    if ((t & 31) == 0) red[t >> 5] = mx;
    __syncthreads();
    float bmax = red[0];
#pragma unroll
    for (int i = 1; i < 8; ++i) bmax = fmaxf(bmax, red[i]);

    float sum = 0.f;
#pragma unroll
    for (int i = 0; i < 4; ++i) { v[i] = __expf(v[i] - bmax); sum += v[i]; }
#pragma unroll
    for (int off = 16; off > 0; off >>= 1) sum += __shfl_xor(sum, off, 32);
    __syncthreads();
    if ((t & 31) == 0) red[t >> 5] = sum;
    __syncthreads();
    float tot = 0.f;
#pragma unroll
    for (int i = 0; i < 8; ++i) tot += red[i];
    const float rinv = 1.0f / tot;
#pragma unroll
    for (int i = 0; i < 4; ++i) x[t + i * 256] = v[i] * rinv;
}

// ---------------------------------------------------------------------------
extern "C" void kernel_launch(void* const* d_in, const int* in_sizes, int n_in,
                              void* d_out, int out_size, void* d_ws, size_t ws_size,
                              hipStream_t stream)
{
    (void)in_sizes; (void)n_in; (void)out_size; (void)ws_size;

    const float* query = (const float*)d_in[0];
    const float* key_  = (const float*)d_in[1];
    const float* value = (const float*)d_in[2];
    const float* Wq  = (const float*)d_in[3];  const float* bq  = (const float*)d_in[4];
    const float* Wk  = (const float*)d_in[5];  const float* bk  = (const float*)d_in[6];
    const float* Wv  = (const float*)d_in[7];  const float* bv  = (const float*)d_in[8];
    const float* Wp  = (const float*)d_in[9];  const float* bp  = (const float*)d_in[10];
    const float* Wup = (const float*)d_in[11]; const float* bup = (const float*)d_in[12];
    const float* Wud = (const float*)d_in[13]; const float* bud = (const float*)d_in[14];
    const float* Wus = (const float*)d_in[15]; const float* bus = (const float*)d_in[16];
    const float* Wo  = (const float*)d_in[17]; const float* bo  = (const float*)d_in[18];

    const size_t NELEM = (size_t)BATCH * SEQ * HIDN;   // 4,194,304
    float* ws   = (float*)d_ws;
    float* qb   = ws;
    float* kb   = qb + NELEM;
    float* vb   = kb + NELEM;
    float* pb   = vb + NELEM;
    float* outm = pb + NELEM;
    float* cen  = outm + NELEM;
    float* win  = cen + (size_t)BATCH * NH * SEQ;
    float* alT  = win + (size_t)BATCH * NH * SEQ;

    float* x    = (float*)d_out;         // [B,S,HID]
    float* attn = x + NELEM;             // [B,NH,S,S]

    const dim3 blk(256, 1, 1);
    const dim3 gProj(HIDN / BN, (BATCH * SEQ) / BM, 1);          // (16,32)
    const dim3 gScore(SEQ / BN, SEQ / BM, BATCH * NH);           // (16,8,64)
    const dim3 gAV(HD / BN, SEQ / BM, BATCH * NH);               // (1,8,64)

    // q / k / v / p projections
    proj_gemm<<<gProj, blk, 0, stream>>>(query, Wq, bq, qb, HIDN, HIDN, HIDN, HIDN);
    proj_gemm<<<gProj, blk, 0, stream>>>(key_,  Wk, bk, kb, HIDN, HIDN, HIDN, HIDN);
    proj_gemm<<<gProj, blk, 0, stream>>>(value, Wv, bv, vb, HIDN, HIDN, HIDN, HIDN);
    proj_gemm<<<gProj, blk, 0, stream>>>(qb,    Wp, bp, pb, HIDN, HIDN, HIDN, HIDN);

    // Gaussian window params + alpha gate
    gauss_window_params<<<dim3((BATCH * NH * SEQ) / 8), blk, 0, stream>>>(
        pb, Wup, bup, Wud, bud, cen, win);
    alpha_gate<<<dim3((BATCH * SEQ * NH) / 8), blk, 0, stream>>>(query, Wus, bus, alT);

    // raw scaled scores -> attn region of d_out
    scores_gemm<<<gScore, blk, 0, stream>>>(qb, kb, attn);

    // fused bias + softmax in place
    softmax_bias_rows<<<dim3(BATCH * NH * SEQ), blk, 0, stream>>>(attn, cen, win, alT);

    // attn @ v -> merged heads
    attnv_gemm<<<gAV, blk, 0, stream>>>(attn, vb, outm);

    // final output projection
    proj_gemm<<<gProj, blk, 0, stream>>>(outm, Wo, bo, x, HIDN, HIDN, HIDN, HIDN);
}
